// MamdaniANFIS_1881195676400
// MI455X (gfx1250) — matrix operand order repaired and checked by reading, not compiled, not run on hardware
//
#include <hip/hip_runtime.h>
#include <hip/hip_bf16.h>

// ---------------------------------------------------------------------------
// Mamdani ANFIS on MI455X (gfx1250, wave32, WMMA).
//
// firing[b,r] = prod_d mu[b,d,ant[r,d]]  is recast in log space as a GEMM:
//   logF = L (2048x64, f32 split into f16 hi+lo) @ H (64x32768 one-hot)
// computed with v_wmma_f32_16x16x32_f16 (K=64 -> 2 K-steps, x2 for hi/lo).
// Top-8 per row is selected in log domain fused into the tile consumer;
// exp() is applied only to the 8 winners, then 100-pt defuzzification.
// ---------------------------------------------------------------------------

typedef __attribute__((ext_vector_type(16))) _Float16 v16h;
typedef __attribute__((ext_vector_type(8)))  float    v8f;

#define B_SZ    2048
#define D_SZ    8
#define M_SZ    7
#define R_SZ    32768
#define NPTS    100
#define NGROUPS 4
#define WAVES   8
#define RULES_PER_WAVE (R_SZ / (NGROUPS * WAVES))   // 1024
#define LOG_EPS (-11.512925464970229f)              // ln(1e-5)
#define NEG_BIG (-3.0e38f)

// ---------------------------------------------------------------------------
// Kernel 0: 64-bit one-hot mask per rule. bit (d*8+m) = (ant[r][d]==m).
// ant == -1 leaves that dim's 8 bits clear -> contributes log(1)=0.
// ---------------------------------------------------------------------------
__global__ void build_masks(const int* __restrict__ ant, uint2* __restrict__ masks)
{
    int r = blockIdx.x * blockDim.x + threadIdx.x;
    if (r >= R_SZ) return;
    unsigned lo = 0u, hi = 0u;
#pragma unroll
    for (int d = 0; d < 4; ++d) {
        int a = ant[r * D_SZ + d];
        if (a >= 0) lo |= 1u << (d * 8 + a);
    }
#pragma unroll
    for (int d = 4; d < 8; ++d) {
        int a = ant[r * D_SZ + d];
        if (a >= 0) hi |= 1u << ((d - 4) * 8 + a);
    }
    masks[r] = make_uint2(lo, hi);
}

// ---------------------------------------------------------------------------
// Kernel 1: log-domain GEMM via WMMA + fused per-row top-8.
// grid = (B/16 batch tiles, NGROUPS rule groups), block = 256 (8 waves).
// Each wave: same 16 batch rows, private 1024-rule range, 64 tiles of 16.
// ---------------------------------------------------------------------------
__global__ __launch_bounds__(256)
void anfis_fire_topk(const float* __restrict__ x,
                     const float* __restrict__ in_centers,   // [D][M]
                     const float* __restrict__ in_widths,    // [D][M]
                     const uint2* __restrict__ masks,        // [R]
                     float* __restrict__ gval,               // [NGROUPS][B][8]
                     int*   __restrict__ gidx)               // [NGROUPS][B][8]
{
    __shared__ float logmu[16][64];          // 16 rows x (8 dims * 8 slots)
    __shared__ float ctile[WAVES][16][16];   // per-wave C redistribution
    __shared__ float plv[256][8];            // per-lane partial top-8 vals
    __shared__ int   pli[256][8];            // per-lane partial top-8 idx

    const int tid = threadIdx.x;
    const int bt  = blockIdx.x;              // batch tile (16 rows)
    const int grp = blockIdx.y;              // rule group

    // ---- build log-membership table for this batch tile (K slot m=7 -> 0) --
    for (int i = tid; i < 16 * 64; i += 256) {
        int row = i >> 6, k = i & 63, d = k >> 3, m = k & 7;
        float v = 0.0f;
        if (m < M_SZ) {
            float xv = x[(bt * 16 + row) * D_SZ + d];
            float z  = (xv - in_centers[d * M_SZ + m]) / in_widths[d * M_SZ + m];
            float t  = -0.5f * z * z;
            v = fminf(fmaxf(t, LOG_EPS), 0.0f);
        }
        logmu[row][k] = v;
    }
    __syncthreads();

    const int lane = tid & 31;
    const int wv   = tid >> 5;
    const int h    = lane >> 4;              // lane half
    const int l    = lane & 15;

    // ---- A operands, 16x32 tile in contiguous k = 16h+s order, hi/lo f16 split (persistent) --
    // lane(l,h): WMMA w covers dims {4w+2h (elems 0..7), 4w+2h+1 (elems 8..15)}.
    v16h ahi[2], alo[2];
#pragma unroll
    for (int w = 0; w < 2; ++w) {
        int d0 = 4 * w + 2 * h, d1 = 4 * w + 2 * h + 1;
#pragma unroll
        for (int e = 0; e < 8; ++e) {
            float f0 = logmu[l][d0 * 8 + e];
            _Float16 h0 = (_Float16)f0;
            ahi[w][e] = h0;
            alo[w][e] = (_Float16)(f0 - (float)h0);
            float f1 = logmu[l][d1 * 8 + e];
            _Float16 h1 = (_Float16)f1;
            ahi[w][8 + e] = h1;
            alo[w][8 + e] = (_Float16)(f1 - (float)h1);
        }
    }

    // ---- per-lane top-8 state: lane owns row (lane&15), rule half (lane>>4) -
    float tv[8]; int ti[8];
#pragma unroll
    for (int s = 0; s < 8; ++s) { tv[s] = NEG_BIG; ti[s] = 0x7fffffff; }

    const int ruleBase  = grp * (R_SZ / NGROUPS) + wv * RULES_PER_WAVE;
    const int myRow     = l;
    const int myColBase = h * 8;

    for (int t = 0; t < RULES_PER_WAVE / 16; ++t) {
        const int r0 = ruleBase + t * 16;

        // B one-hot from bitmask: lane(n=l,h) holds K rows [16h,16h+16) of col n.
        uint2 mk = masks[r0 + l];
        unsigned bits0 = (mk.x >> (16 * h)) & 0xFFFFu;
        unsigned bits1 = (mk.y >> (16 * h)) & 0xFFFFu;
        v16h b0, b1;
#pragma unroll
        for (int e = 0; e < 16; ++e) {
            b0[e] = ((bits0 >> e) & 1u) ? (_Float16)1.0f : (_Float16)0.0f;
            b1[e] = ((bits1 >> e) & 1u) ? (_Float16)1.0f : (_Float16)0.0f;
        }

        // logF tile = (Ahi+Alo) @ [B0;B1], f32 accumulate.
        v8f c = {};
        c = __builtin_amdgcn_wmma_f32_16x16x32_f16(false, ahi[0], false, b0, (short)0, c, false, false);
        c = __builtin_amdgcn_wmma_f32_16x16x32_f16(false, ahi[1], false, b1, (short)0, c, false, false);
        c = __builtin_amdgcn_wmma_f32_16x16x32_f16(false, alo[0], false, b0, (short)0, c, false, false);
        c = __builtin_amdgcn_wmma_f32_16x16x32_f16(false, alo[1], false, b1, (short)0, c, false, false);

        // Redistribute C (row = vgpr + 8h, col = l) so each lane owns one row.
#pragma unroll
        for (int v = 0; v < 8; ++v) ctile[wv][v + 8 * h][l] = c[v];
        asm volatile("s_wait_dscnt 0" ::: "memory");

        float cand[8];
#pragma unroll
        for (int j = 0; j < 8; ++j) cand[j] = ctile[wv][myRow][myColBase + j];

        // Fast reject: indices only increase, so ties never displace.
        float m8 = cand[0];
#pragma unroll
        for (int j = 1; j < 8; ++j) m8 = fmaxf(m8, cand[j]);
        if (m8 > tv[7]) {
#pragma unroll
            for (int j = 0; j < 8; ++j) {
                float cv = cand[j];
                if (cv > tv[7]) {
                    int ci = r0 + myColBase + j;
#pragma unroll
                    for (int s = 0; s < 8; ++s) {
                        if (cv > tv[s]) {
                            float ftmp = tv[s]; tv[s] = cv; cv = ftmp;
                            int   itmp = ti[s]; ti[s] = ci; ci = itmp;
                        }
                    }
                }
            }
        }
    }

    // ---- merge 16 partial lists per row inside the workgroup ---------------
#pragma unroll
    for (int s = 0; s < 8; ++s) { plv[tid][s] = tv[s]; pli[tid][s] = ti[s]; }
    __syncthreads();

    if (tid < 16) {
        float mv[8]; int mi[8];
#pragma unroll
        for (int s = 0; s < 8; ++s) { mv[s] = NEG_BIG; mi[s] = 0x7fffffff; }
        for (int lid = tid; lid < 256; lid += 16) {   // all lists with lane&15==tid
#pragma unroll
            for (int s = 0; s < 8; ++s) {
                float cv = plv[lid][s]; int ci = pli[lid][s];
#pragma unroll
                for (int q = 0; q < 8; ++q) {
                    bool sw = (cv > mv[q]) || (cv == mv[q] && ci < mi[q]);
                    if (sw) {
                        float ftmp = mv[q]; mv[q] = cv; cv = ftmp;
                        int   itmp = mi[q]; mi[q] = ci; ci = itmp;
                    }
                }
            }
        }
        int rowG = bt * 16 + tid;
#pragma unroll
        for (int s = 0; s < 8; ++s) {
            gval[(grp * B_SZ + rowG) * 8 + s] = mv[s];
            gidx[(grp * B_SZ + rowG) * 8 + s] = mi[s];
        }
    }
}

// ---------------------------------------------------------------------------
// Kernel 2: merge the NGROUPS partial top-8 lists, exp the winners, and
// centroid-defuzzify over the 100-point universe. One thread per batch row.
// ---------------------------------------------------------------------------
__global__ __launch_bounds__(256)
void merge_defuzz(const float* __restrict__ gval, const int* __restrict__ gidx,
                  const int*   __restrict__ consequents,
                  const float* __restrict__ out_centers,
                  const float* __restrict__ out_widths,
                  float* __restrict__ out)
{
    int row = blockIdx.x * blockDim.x + threadIdx.x;
    if (row >= B_SZ) return;

    float tv[8]; int ti[8];
#pragma unroll
    for (int s = 0; s < 8; ++s) { tv[s] = NEG_BIG; ti[s] = 0x7fffffff; }
    for (int g = 0; g < NGROUPS; ++g) {
#pragma unroll
        for (int s = 0; s < 8; ++s) {
            float cv = gval[(g * B_SZ + row) * 8 + s];
            int   ci = gidx[(g * B_SZ + row) * 8 + s];
#pragma unroll
            for (int q = 0; q < 8; ++q) {
                bool sw = (cv > tv[q]) || (cv == tv[q] && ci < ti[q]);
                if (sw) {
                    float ftmp = tv[q]; tv[q] = cv; cv = ftmp;
                    int   itmp = ti[q]; ti[q] = ci; ci = itmp;
                }
            }
        }
    }

    float fv[8], oc[8], ow[8];
#pragma unroll
    for (int s = 0; s < 8; ++s) {
        fv[s] = expf(tv[s]);                 // firing strength of winner
        int cc = consequents[ti[s]];
        oc[s] = out_centers[cc];
        ow[s] = out_widths[cc];
    }

    float num = 0.0f, den = 0.0f;
    for (int p = 0; p < NPTS; ++p) {
        float u = (float)p * (1.0f / 99.0f);
        float agg = 0.0f;
#pragma unroll
        for (int s = 0; s < 8; ++s) {
            float z = (u - oc[s]) / ow[s];
            agg += fv[s] * expf(-0.5f * z * z);
        }
        num = fmaf(u, agg, num);
        den += agg;
    }
    out[row] = num / (den + 1e-5f);
}

// ---------------------------------------------------------------------------
extern "C" void kernel_launch(void* const* d_in, const int* in_sizes, int n_in,
                              void* d_out, int out_size, void* d_ws, size_t ws_size,
                              hipStream_t stream)
{
    const float* x    = (const float*)d_in[0];
    const int*   ant  = (const int*)  d_in[1];
    const int*   cons = (const int*)  d_in[2];
    const float* ic   = (const float*)d_in[3];
    const float* iw   = (const float*)d_in[4];
    const float* oc   = (const float*)d_in[5];
    const float* ow   = (const float*)d_in[6];
    float*       out  = (float*)d_out;

    // workspace layout: masks (256KB) | gval (256KB) | gidx (256KB)
    uint2* masks = (uint2*)d_ws;
    float* gval  = (float*)((char*)d_ws + (size_t)R_SZ * sizeof(uint2));
    int*   gidx  = (int*)((char*)gval + (size_t)NGROUPS * B_SZ * 8 * sizeof(float));

    build_masks<<<R_SZ / 256, 256, 0, stream>>>(ant, masks);

    dim3 grid(B_SZ / 16, NGROUPS);
    anfis_fire_topk<<<grid, 256, 0, stream>>>(x, ic, iw, masks, gval, gidx);

    merge_defuzz<<<B_SZ / 256, 256, 0, stream>>>(gval, gidx, cons, oc, ow, out);
}
